// StrataLayer_49443663512015
// MI455X (gfx1250) — matrix operand order repaired and checked by reading, not compiled, not run on hardware
//
#include <hip/hip_runtime.h>

// ---------------- types & constants ----------------
typedef __attribute__((ext_vector_type(16))) __bf16 v16bf;
typedef __attribute__((ext_vector_type(8)))  float  v8f;
typedef __attribute__((ext_vector_type(4)))  unsigned int u32x4;
typedef __attribute__((ext_vector_type(8)))  int i32x8;
typedef __attribute__((ext_vector_type(4)))  int i32x4;

#define NSEQ   32      // number of windows (N)
#define WMAX   128     // max window width
#define BQ     4       // batch
#define DMODEL 512
#define FDIM   2048
#define NHEADS 8
#define DH     64
#define LENGTH 512
#define HALF   64
#define MTOK   (NSEQ * WMAX * BQ)   // 16384 token rows

// ---------------- small device helpers ----------------
__device__ __forceinline__ unsigned short f2bf(float f) {
  unsigned int u = __float_as_uint(f);
  u += 0x7FFFu + ((u >> 16) & 1u);      // round-to-nearest-even
  return (unsigned short)(u >> 16);
}
__device__ __forceinline__ v8f zero8() {
  v8f z = {0.f, 0.f, 0.f, 0.f, 0.f, 0.f, 0.f, 0.f};
  return z;
}
__device__ __forceinline__ v8f wmma_bf16(v16bf a, v16bf b, v8f c) {
  return __builtin_amdgcn_wmma_f32_16x16x32_bf16(false, a, false, b, (short)0, c, false, false);
}

// A-matrix fragment (16x32 bf16): lanes 0-15 -> M=lane, K in {0..7,16..23}+kBase
//                                 lanes 16-31 -> M=lane-16, K in {8..15,24..31}+kBase
__device__ __forceinline__ v16bf load_a_frag(const unsigned short* base, int stride,
                                             int mBase, int kBase, int lane) {
  union { v16bf v; unsigned int u[8]; } f;
  const unsigned short* row = base + (size_t)(mBase + (lane & 15)) * stride;
  const int kb = kBase + ((lane >> 4) << 3);
#pragma unroll
  for (int j = 0; j < 8; ++j) {
    int k = kb + ((j < 4) ? (2 * j) : (2 * j + 8));
    f.u[j] = *(const unsigned int*)(row + k);
  }
  return f.v;
}
// B-matrix fragment (32x16 bf16) from n-major LDS ([n][k]):
//   lanes 0-15 -> N=lane, K in {0..7,16..23}+kBase ; lanes 16-31 -> N=lane-16, K in {8..15,24..31}+kBase
__device__ __forceinline__ v16bf load_b_frag(const unsigned short* base, int stride,
                                             int nBase, int kBase, int lane) {
  union { v16bf v; unsigned int u[8]; } f;
  const unsigned short* row = base + (size_t)(nBase + (lane & 15)) * stride;
  const int kb = kBase + ((lane >> 4) << 3);
#pragma unroll
  for (int j = 0; j < 8; ++j) f.u[j] = *(const unsigned int*)(row + kb + ((j < 4) ? (2 * j) : (2 * j + 8)));
  return f.v;
}

// replicate host-side window math exactly
__device__ __forceinline__ void window_params(int i, int& low, int& high, int& entry, int& Wn) {
  double p = (double)i * (double)(LENGTH - 1) / (double)(NSEQ - 1);
  int ip = (int)p;
  low  = ip - HALF; if (low < 0) low = 0;
  high = ip + HALF; if (high > LENGTH) high = LENGTH;
  int e = (int)ceil(p - (double)HALF); if (e < 0) e = 0;
  entry = e;
  Wn = high - low;
}

// ---------------- TDM: load a 64(N) x 32(K) bf16 tile of WT[N][K] into LDS ----------------
// LDS rows padded to 34 ushorts (68B = 17 DW): pad_interval=16DW, pad_amount=1DW.
__device__ __forceinline__ void tdm_load_b_tile(const unsigned short* WT, int Ntot, int Ktot,
                                                int bn, int k0, unsigned short* ldsDst) {
  unsigned long long ga = (unsigned long long)(const void*)(WT + (size_t)bn * Ktot + k0);
  unsigned int lds = (unsigned int)(unsigned long long)(const void*)ldsDst;
  u32x4 g0;
  g0.x = 1u;                                               // count=1 (valid descriptor)
  g0.y = lds;                                              // lds_addr
  g0.z = (unsigned int)(ga & 0xffffffffull);               // global_addr[31:0]
  g0.w = (unsigned int)((ga >> 32) & 0x01ffffffull)        // global_addr[56:32]
       | (2u << 30);                                       // type=2 ("image")
  i32x8 g1;
  g1[0] = (int)(0x00010000u       // data_size=1 -> 2 bytes
              | 0x00100000u       // pad_enable
              | (3u << 22));      // pad_interval code 3 -> every 16 DWORDs
                                  // pad_amount code 0   -> 1 DWORD
  g1[1] = (int)((unsigned)Ktot << 16);                               // tensor_dim0[15:0]
  g1[2] = (int)(((unsigned)Ktot >> 16) | ((unsigned)Ntot << 16));    // dim0 hi | dim1 lo
  g1[3] = (int)(((unsigned)Ntot >> 16) | (32u << 16));               // dim1 hi | tile_dim0=32
  g1[4] = 64;                                                        // tile_dim1=64
  g1[5] = (int)(unsigned)Ktot;                                       // tensor_dim0_stride lo
  g1[6] = 0;
  g1[7] = 0;
  i32x4 gz = {0, 0, 0, 0};                                           // 2D: groups 2/3 unused
  i32x8 gz8 = {0, 0, 0, 0, 0, 0, 0, 0};
  __builtin_amdgcn_tensor_load_to_lds(g0, g1, gz, gz, gz8, 0);
}

// ---------------- weight pre-conversion: W[K][N] fp32 -> WT[N][K] bf16 ----------------
__global__ __launch_bounds__(256) void convert_wt_kernel(const float* __restrict__ W,
                                                         unsigned short* __restrict__ WT,
                                                         int K, int N) {
  int idx = blockIdx.x * 256 + threadIdx.x;
  int total = K * N;
  if (idx < total) {
    int n = idx / K, k = idx - n * K;
    WT[idx] = f2bf(W[(size_t)k * N + n]);
  }
}

// ---------------- positional encoding + input broadcast ----------------
__global__ __launch_bounds__(256) void pe_add_kernel(const float* __restrict__ In,
                                                     float* __restrict__ X) {
  const int iw = blockIdx.x;             // 0..NSEQ*WMAX-1
  const int i = iw >> 7, w = iw & 127;
  int low, high, entry, Wn;
  window_params(i, low, high, entry, Wn);
  const float pos = (float)(low + w);
  const int tid = threadIdx.x;
#pragma unroll
  for (int rep = 0; rep < 2; ++rep) {
    int d = tid + rep * 256;
    float j2 = (float)((d >> 1) << 1);
    float dv = __expf(j2 * (-9.21034037198f / (float)DMODEL));   // ln(10000)
    float ang = pos * dv;
    float pe = (d & 1) ? cosf(ang) : sinf(ang);
#pragma unroll
    for (int b = 0; b < BQ; ++b) {
      X[(size_t)(((i * WMAX + w) * BQ + b)) * DMODEL + d] =
          In[(size_t)(i * BQ + b) * DMODEL + d] + pe;
    }
  }
}

// ---------------- bf16 WMMA GEMM: C = A[MxK] * W[KxN] + bias (opt relu) ----------------
// A: fp32 activations (converted to bf16 per-tile in LDS).
// WT: bf16 transposed weights [N][K], staged to LDS by the Tensor Data Mover.
// Double-buffered K-loop: TDM + A-conversion for tile k+1 overlap WMMA on tile k.
__device__ __forceinline__ void store_tile(float* __restrict__ C, int N, v8f a,
                                           int m0, int n0, const float* __restrict__ bias,
                                           int relu, int lane) {
  const int nl = lane & 15;
  const int mh = (lane >> 4) << 3;
  const int n = n0 + nl;
  const float bv = bias[n];
#pragma unroll
  for (int v = 0; v < 8; ++v) {
    float val = a[v] + bv;
    if (relu) val = fmaxf(val, 0.f);
    C[(size_t)(m0 + v + mh) * N + n] = val;
  }
}

__global__ __launch_bounds__(128) void gemm_wmma_kernel(const float* __restrict__ A,
                                                        const unsigned short* __restrict__ WT,
                                                        const float* __restrict__ bias,
                                                        float* __restrict__ C,
                                                        int M, int N, int K, int relu) {
  const int tid  = threadIdx.x;
  const int lane = tid & 31;
  const int wid  = tid >> 5;
  const int bm = blockIdx.y << 6;
  const int bn = blockIdx.x << 6;
  const int wr = (wid >> 1) << 5;    // wave row inside 64x64 tile
  const int wc = (wid & 1)  << 5;    // wave col

  __shared__ __align__(16) unsigned short As[2][64 * 34];  // [m][k] bf16
  __shared__ __align__(16) unsigned short Bs[2][64 * 34];  // [n][k] bf16 (n-major, TDM-filled)

  v8f acc00 = zero8(), acc01 = zero8(), acc10 = zero8(), acc11 = zero8();

  // prologue: stage tile k0=0 into buffer 0
  if (wid == 0) tdm_load_b_tile(WT, N, K, bn, 0, &Bs[0][0]);
#pragma unroll
  for (int l = tid; l < 2048; l += 128) {
    int r = l >> 5, c = l & 31;
    As[0][r * 34 + c] = f2bf(A[(size_t)(bm + r) * K + c]);
  }
  if (wid == 0) __builtin_amdgcn_s_wait_tensorcnt((short)0);
  __syncthreads();

  int cur = 0;
  for (int k0 = 0; k0 < K; k0 += 32) {
    const int nxt = k0 + 32;
    if (nxt < K) {
      // prefetch next tile into the other buffer while this one is consumed
      if (wid == 0) tdm_load_b_tile(WT, N, K, bn, nxt, &Bs[cur ^ 1][0]);
#pragma unroll
      for (int l = tid; l < 2048; l += 128) {
        int r = l >> 5, c = l & 31;
        As[cur ^ 1][r * 34 + c] = f2bf(A[(size_t)(bm + r) * K + nxt + c]);
      }
    }

    v16bf a0 = load_a_frag(&As[cur][0], 34, wr,      0, lane);
    v16bf a1 = load_a_frag(&As[cur][0], 34, wr + 16, 0, lane);
    v16bf b0 = load_b_frag(&Bs[cur][0], 34, wc,      0, lane);
    v16bf b1 = load_b_frag(&Bs[cur][0], 34, wc + 16, 0, lane);
    acc00 = wmma_bf16(a0, b0, acc00);
    acc01 = wmma_bf16(a0, b1, acc01);
    acc10 = wmma_bf16(a1, b0, acc10);
    acc11 = wmma_bf16(a1, b1, acc11);

    if (wid == 0 && nxt < K) __builtin_amdgcn_s_wait_tensorcnt((short)0);
    __syncthreads();
    cur ^= 1;
  }
  store_tile(C, N, acc00, bm + wr,      bn + wc,      bias, relu, lane);
  store_tile(C, N, acc01, bm + wr,      bn + wc + 16, bias, relu, lane);
  store_tile(C, N, acc10, bm + wr + 16, bn + wc,      bias, relu, lane);
  store_tile(C, N, acc11, bm + wr + 16, bn + wc + 16, bias, relu, lane);
}

// ---------------- attention: one block per (i,b,h), 4 waves ----------------
__global__ __launch_bounds__(128) void attn_wmma_kernel(const float* __restrict__ Q,
                                                        const float* __restrict__ K,
                                                        const float* __restrict__ V,
                                                        float* __restrict__ O) {
  const int h = blockIdx.x, b = blockIdx.y, iSeq = blockIdx.z;
  const int tid = threadIdx.x, lane = tid & 31, wid = tid >> 5;

  __shared__ __align__(16) unsigned short Qs[128 * 66];   // [w][d]
  __shared__ __align__(16) unsigned short Ks[128 * 66];   // [key][d] (n-major for S)
  __shared__ __align__(16) unsigned short Vs[64 * 130];   // [d][key] (n-major for O)
  __shared__ __align__(16) float          Ss[128 * 130];  // scores fp32; P bf16 aliases rows

  int low, high, entry, Wn;
  window_params(iSeq, low, high, entry, Wn);

  const size_t base = (size_t)iSeq * (WMAX * BQ * DMODEL) + (size_t)b * DMODEL + (size_t)h * DH;
  const size_t wstride = (size_t)BQ * DMODEL;

#pragma unroll
  for (int l = tid; l < 128 * 64; l += 128) {
    int w = l >> 6, d = l & 63;
    size_t g = base + (size_t)w * wstride + d;
    Qs[w * 66 + d]  = f2bf(Q[g]);
    Ks[w * 66 + d]  = f2bf(K[g]);
    Vs[d * 130 + w] = f2bf(V[g]);
  }
  __syncthreads();

  // S = (Q K^T) / sqrt(dh)   M=128 N=128 K=64
#pragma unroll
  for (int mt = 0; mt < 2; ++mt) {
    const int m0 = wid * 32 + mt * 16;
    v16bf a0 = load_a_frag(Qs, 66, m0, 0, lane);
    v16bf a1 = load_a_frag(Qs, 66, m0, 32, lane);
#pragma unroll
    for (int nt = 0; nt < 8; ++nt) {
      v8f acc = zero8();
      v16bf b0 = load_b_frag(Ks, 66, nt * 16, 0, lane);
      v16bf b1 = load_b_frag(Ks, 66, nt * 16, 32, lane);
      acc = wmma_bf16(a0, b0, acc);
      acc = wmma_bf16(a1, b1, acc);
      const int nl = nt * 16 + (lane & 15);
      const int mh = (lane >> 4) << 3;
#pragma unroll
      for (int v = 0; v < 8; ++v)
        Ss[(m0 + v + mh) * 130 + nl] = acc[v] * 0.125f;
    }
  }
  __syncthreads();

  // masked softmax per query row (one thread per row); P(bf16) overwrites S row in place
  {
    const int w = tid;
    float* srow = Ss + (size_t)w * 130;
    unsigned short* prow = (unsigned short*)srow;
    if (w < Wn) {
      float mx = -3.4e38f;
      for (int c = 0; c < Wn; ++c) mx = fmaxf(mx, srow[c]);
      float sum = 0.f;
      for (int c = 0; c < Wn; ++c) sum += __expf(srow[c] - mx);
      const float inv = 1.f / sum;
      for (int c = 0; c < 128; ++c) {
        float e = (c < Wn) ? (__expf(srow[c] - mx) * inv) : 0.f;
        prow[c] = f2bf(e);   // safe: byte 2c < byte 4c of the float just read
      }
    } else {
      for (int c = 0; c < 128; ++c) prow[c] = 0;  // padded query rows -> zero output
    }
  }
  __syncthreads();

  // O = P V   M=128 N=64 K=128
#pragma unroll
  for (int mt = 0; mt < 2; ++mt) {
    const int m0 = wid * 32 + mt * 16;
#pragma unroll
    for (int nt = 0; nt < 4; ++nt) {
      v8f acc = zero8();
#pragma unroll
      for (int kk = 0; kk < 128; kk += 32) {
        v16bf a  = load_a_frag((const unsigned short*)Ss, 260, m0, kk, lane);
        v16bf bb = load_b_frag(Vs, 130, nt * 16, kk, lane);
        acc = wmma_bf16(a, bb, acc);
      }
      const int d  = nt * 16 + (lane & 15);
      const int mh = (lane >> 4) << 3;
#pragma unroll
      for (int v = 0; v < 8; ++v) {
        int w = m0 + v + mh;
        O[base + (size_t)w * wstride + d] = acc[v];
      }
    }
  }
}

// ---------------- fused residual + LayerNorm (one block per token row) ----------------
__global__ __launch_bounds__(256) void ln_residual_kernel(const float* __restrict__ X,
                                                          const float* __restrict__ R,
                                                          const float* __restrict__ g,
                                                          const float* __restrict__ be,
                                                          float* __restrict__ Y) {
  const int row = blockIdx.x;
  const int tid = threadIdx.x;
  const size_t roff = (size_t)row * DMODEL;
  float v0 = X[roff + tid]       + R[roff + tid];
  float v1 = X[roff + tid + 256] + R[roff + tid + 256];
  float s = v0 + v1, q = v0 * v0 + v1 * v1;
  __shared__ float rs[256];
  __shared__ float rq[256];
  rs[tid] = s; rq[tid] = q;
  __syncthreads();
  for (int off = 128; off > 0; off >>= 1) {
    if (tid < off) { rs[tid] += rs[tid + off]; rq[tid] += rq[tid + off]; }
    __syncthreads();
  }
  const float mean = rs[0] * (1.f / (float)DMODEL);
  const float var  = rq[0] * (1.f / (float)DMODEL) - mean * mean;
  const float inv  = rsqrtf(var + 1e-5f);
  Y[roff + tid]       = (v0 - mean) * inv * g[tid]       + be[tid];
  Y[roff + tid + 256] = (v1 - mean) * inv * g[tid + 256] + be[tid + 256];
}

// ---------------- final Gaussian-softmax gather ----------------
__global__ __launch_bounds__(256) void final_gather_kernel(const float* __restrict__ Xf,
                                                           const float* __restrict__ Vert,
                                                           const float* __restrict__ Horz,
                                                           float* __restrict__ Out) {
  const int t = blockIdx.x, b = blockIdx.y, tid = threadIdx.x;
  const float hscale = (float)LENGTH / (float)NSEQ;
#pragma unroll
  for (int rep = 0; rep < 2; ++rep) {
    const int d = tid + rep * 256;
    float logit[NSEQ];
    float mx = -3.4e38f;
#pragma unroll
    for (int i = 0; i < NSEQ; ++i) {
      int low, high, entry, Wn;
      window_params(i, low, high, entry, Wn);
      const bool active = (entry <= t) && (t <= high);
      const float v  = Vert[(size_t)(i * BQ + b) * DMODEL + d];
      const float hh = hscale * Horz[(size_t)(i * BQ + b) * DMODEL + d];
      float l = -3.4e38f;
      if (active && v > 0.f) {
        float dm = (float)t - (float)(high - HALF);
        l = v * __expf(-(dm * dm) / (2.f * hh * hh + 1e-10f));
      }
      logit[i] = l;
      mx = fmaxf(mx, l);
    }
    float sum = 0.f, acc = 0.f;
#pragma unroll
    for (int i = 0; i < NSEQ; ++i) {
      if (logit[i] > -1e37f) {
        float e = __expf(logit[i] - mx);
        sum += e;
        int low, high, entry, Wn;
        window_params(i, low, high, entry, Wn);
        int gidx = (t - low) % Wn;
        acc += e * Xf[(size_t)(((i * WMAX + gidx) * BQ + b)) * DMODEL + d];
      }
    }
    Out[(size_t)(t * BQ + b) * DMODEL + d] = acc / sum;
  }
}

// ---------------- host orchestration ----------------
struct LayerPtrs {
  const float *b1,*b2,*bk,*bo,*bq,*bv,*ln1_b,*ln1_g,*ln2_b,*ln2_g,*w1,*w2,*wk,*wo,*wq,*wv;
};
struct LayerW {           // transposed bf16 weights in workspace
  unsigned short *wq,*wk,*wv,*wo,*w1,*w2;
};
// jax pytree flatten order (dict keys sorted alphabetically)
static LayerPtrs layer_from(void* const* d_in, int base) {
  LayerPtrs L;
  L.b1    = (const float*)d_in[base + 0];
  L.b2    = (const float*)d_in[base + 1];
  L.bk    = (const float*)d_in[base + 2];
  L.bo    = (const float*)d_in[base + 3];
  L.bq    = (const float*)d_in[base + 4];
  L.bv    = (const float*)d_in[base + 5];
  L.ln1_b = (const float*)d_in[base + 6];
  L.ln1_g = (const float*)d_in[base + 7];
  L.ln2_b = (const float*)d_in[base + 8];
  L.ln2_g = (const float*)d_in[base + 9];
  L.w1    = (const float*)d_in[base + 10];
  L.w2    = (const float*)d_in[base + 11];
  L.wk    = (const float*)d_in[base + 12];
  L.wo    = (const float*)d_in[base + 13];
  L.wq    = (const float*)d_in[base + 14];
  L.wv    = (const float*)d_in[base + 15];
  return L;
}

static void convert_layer_weights(const LayerPtrs& P, const LayerW& Wt, hipStream_t s) {
  const int DD = DMODEL * DMODEL, DF = DMODEL * FDIM;
  convert_wt_kernel<<<(DD + 255) / 256, 256, 0, s>>>(P.wq, Wt.wq, DMODEL, DMODEL);
  convert_wt_kernel<<<(DD + 255) / 256, 256, 0, s>>>(P.wk, Wt.wk, DMODEL, DMODEL);
  convert_wt_kernel<<<(DD + 255) / 256, 256, 0, s>>>(P.wv, Wt.wv, DMODEL, DMODEL);
  convert_wt_kernel<<<(DD + 255) / 256, 256, 0, s>>>(P.wo, Wt.wo, DMODEL, DMODEL);
  convert_wt_kernel<<<(DF + 255) / 256, 256, 0, s>>>(P.w1, Wt.w1, DMODEL, FDIM);
  convert_wt_kernel<<<(DF + 255) / 256, 256, 0, s>>>(P.w2, Wt.w2, FDIM, DMODEL);
}

static void run_encoder_layer(const LayerPtrs& P, const LayerW& Wt, float* X, float* X2,
                              float* Qb, float* Kb, float* Vb, float* T0, float* Hb,
                              hipStream_t s) {
  dim3 blk(128);
  dim3 gDD(DMODEL / 64, MTOK / 64);
  dim3 gDF(FDIM  / 64, MTOK / 64);
  gemm_wmma_kernel<<<gDD, blk, 0, s>>>(X,  Wt.wq, P.bq, Qb, MTOK, DMODEL, DMODEL, 0);
  gemm_wmma_kernel<<<gDD, blk, 0, s>>>(X,  Wt.wk, P.bk, Kb, MTOK, DMODEL, DMODEL, 0);
  gemm_wmma_kernel<<<gDD, blk, 0, s>>>(X,  Wt.wv, P.bv, Vb, MTOK, DMODEL, DMODEL, 0);
  attn_wmma_kernel<<<dim3(NHEADS, BQ, NSEQ), dim3(128), 0, s>>>(Qb, Kb, Vb, T0);
  gemm_wmma_kernel<<<gDD, blk, 0, s>>>(T0, Wt.wo, P.bo, X2, MTOK, DMODEL, DMODEL, 0);
  ln_residual_kernel<<<MTOK, 256, 0, s>>>(X, X2, P.ln1_g, P.ln1_b, X);
  gemm_wmma_kernel<<<gDF, blk, 0, s>>>(X,  Wt.w1, P.b1, Hb, MTOK, FDIM, DMODEL, 1);
  gemm_wmma_kernel<<<gDD, blk, 0, s>>>(Hb, Wt.w2, P.b2, X2, MTOK, DMODEL, FDIM, 0);
  ln_residual_kernel<<<MTOK, 256, 0, s>>>(X, X2, P.ln2_g, P.ln2_b, X);
}

extern "C" void kernel_launch(void* const* d_in, const int* in_sizes, int n_in,
                              void* d_out, int out_size, void* d_ws, size_t ws_size,
                              hipStream_t stream) {
  (void)in_sizes; (void)n_in; (void)out_size; (void)ws_size;

  const float* inputs     = (const float*)d_in[0];
  const float* vertical   = (const float*)d_in[1];
  const float* horizontal = (const float*)d_in[2];
  const float* in_b       = (const float*)d_in[3];
  LayerPtrs Lin = layer_from(d_in, 4);
  const float* in_w       = (const float*)d_in[20];
  LayerPtrs L0  = layer_from(d_in, 21);
  LayerPtrs L1  = layer_from(d_in, 37);

  const size_t MD = (size_t)MTOK * DMODEL;
  const size_t DD = (size_t)DMODEL * DMODEL;
  const size_t DF = (size_t)DMODEL * FDIM;

  float* ws = (float*)d_ws;
  float* X  = ws;            // activations
  float* X2 = X  + MD;       // matmul output / residual partner
  float* Qb = X2 + MD;
  float* Kb = Qb + MD;
  float* Vb = Kb + MD;
  float* T0 = Vb + MD;
  float* Hb = Qb;            // FFN hidden (M x F) aliases Qb..T0 (dead by then)

  // bf16 transposed weight area after the fp32 buffers
  unsigned short* wb = (unsigned short*)(T0 + MD);
  unsigned short* in_w_t = wb;  wb += DD;
  LayerW Wt[3];
  for (int l = 0; l < 3; ++l) {
    Wt[l].wq = wb; wb += DD;
    Wt[l].wk = wb; wb += DD;
    Wt[l].wv = wb; wb += DD;
    Wt[l].wo = wb; wb += DD;
    Wt[l].w1 = wb; wb += DF;
    Wt[l].w2 = wb; wb += DF;
  }

  // one-time (per launch) weight conversion to transposed bf16
  convert_wt_kernel<<<(int)((DD + 255) / 256), 256, 0, stream>>>(in_w, in_w_t, DMODEL, DMODEL);
  convert_layer_weights(Lin, Wt[0], stream);
  convert_layer_weights(L0,  Wt[1], stream);
  convert_layer_weights(L1,  Wt[2], stream);

  pe_add_kernel<<<dim3(NSEQ * WMAX), dim3(256), 0, stream>>>(inputs, X);

  run_encoder_layer(Lin, Wt[0], X, X2, Qb, Kb, Vb, T0, Hb, stream);

  gemm_wmma_kernel<<<dim3(DMODEL / 64, MTOK / 64), dim3(128), 0, stream>>>(
      X, in_w_t, in_b, X2, MTOK, DMODEL, DMODEL, 0);
  { float* t = X; X = X2; X2 = t; }

  run_encoder_layer(L0, Wt[1], X, X2, Qb, Kb, Vb, T0, Hb, stream);
  run_encoder_layer(L1, Wt[2], X, X2, Qb, Kb, Vb, T0, Hb, stream);

  final_gather_kernel<<<dim3(LENGTH, BQ), dim3(256), 0, stream>>>(
      X, vertical, horizontal, (float*)d_out);
}
